// DenseGCNConv_58411555225967
// MI455X (gfx1250) — compile-verified, matrix operand order checked by reading, not run on hardware
//
#include <hip/hip_runtime.h>

#define NNODES 16384
#define FDIM   128

typedef __attribute__((ext_vector_type(16))) _Float16 v16h;
typedef __attribute__((ext_vector_type(8)))  _Float16 v8h;
typedef __attribute__((ext_vector_type(8)))  float    v8f;
typedef __attribute__((ext_vector_type(4)))  float    v4f;

union BFrag { v16h v; v8h h[2]; };

// ---------------------------------------------------------------------------
// Stage 1 (off critical path): Pt[fo][node] = dot(features[node], W[fo]) in
// fp32, stored transposed as fp16 so the main GEMM's B-fragments are direct
// contiguous global_load_b128's. 0.54 GFLOP total — negligible.
// ---------------------------------------------------------------------------
__global__ __launch_bounds__(FDIM) void gcn_proj_kernel(
    const float* __restrict__ feat, const float* __restrict__ W,
    _Float16* __restrict__ Pt)
{
  __shared__ float fs[FDIM];
  const int node = blockIdx.x;
  const int t    = threadIdx.x;            // t == output feature index
  fs[t] = feat[(size_t)node * FDIM + t];
  __syncthreads();

  const float* wr = W + (size_t)t * FDIM;  // W row-major [F_OUT][F_IN]
  float acc = 0.0f;
#pragma unroll 16
  for (int k = 0; k < FDIM; ++k)
    acc = __builtin_fmaf(fs[k], wr[k], acc);

  Pt[(size_t)t * NNODES + node] = (_Float16)acc;
}

// ---------------------------------------------------------------------------
// A fragment pack: 16 fp32 -> v16h in the documented 16-bit A 16x32 layout.
// ---------------------------------------------------------------------------
__device__ __forceinline__ v16h cvt_afrag(v4f a0, v4f a1, v4f a2, v4f a3) {
  v16h r;
#pragma unroll
  for (int e = 0; e < 4; ++e) {
    r[e]      = (_Float16)a0[e];
    r[4 + e]  = (_Float16)a1[e];
    r[8 + e]  = (_Float16)a2[e];
    r[12 + e] = (_Float16)a3[e];
  }
  return r;
}

// ---------------------------------------------------------------------------
// One K-chunk step, software-pipelined one chunk ahead:
//   - issue next chunk's A loads (non-temporal: adj streamed exactly once)
//   - 8 WMMAs on current fragments; after consuming bf[j], immediately issue
//     bf[j]'s loads for the next chunk (latency covered by following WMMAs)
//   - convert next A raw into the *other* A buffer (no WAR with live WMMAs)
// ---------------------------------------------------------------------------
__device__ __forceinline__ void gemm_step(
    const float* __restrict__ arow, const _Float16* const* bbase,
    int kn, int khalf, v8f acc[8], const v16h& af_cur, v16h& af_next,
    BFrag bf[8])
{
  v4f a0 = __builtin_nontemporal_load((const v4f*)(arow + kn + khalf));
  v4f a1 = __builtin_nontemporal_load((const v4f*)(arow + kn + khalf + 4));
  v4f a2 = __builtin_nontemporal_load((const v4f*)(arow + kn + khalf + 16));
  v4f a3 = __builtin_nontemporal_load((const v4f*)(arow + kn + khalf + 20));
  __builtin_prefetch(arow + kn + 256, 0, 1);   // global_prefetch_b8, ~1KB ahead

#pragma unroll
  for (int j = 0; j < 8; ++j) {
    acc[j] = __builtin_amdgcn_wmma_f32_16x16x32_f16(
        /*neg_a=*/false, af_cur, /*neg_b=*/false, bf[j].v,
        /*c_mod=*/(short)0, acc[j], /*reuse_a=*/false, /*reuse_b=*/false);
    bf[j].h[0] = *(const v8h*)(bbase[j] + kn);        // K = khalf + 0..7
    bf[j].h[1] = *(const v8h*)(bbase[j] + kn + 16);   // K = 16 + khalf + 0..7
  }
  af_next = cvt_afrag(a0, a1, a2, a3);
}

// ---------------------------------------------------------------------------
// Stage 2 (hot): out[64 x 128] tile per block = adj[64 x 16384] @ Pt^T + b.
// 4 waves / block, each wave: 16-row strip x 8 col-tiles of 16.
// fp16 WMMA 16x16x32, fp32 accumulate. HBM-bound: 1 GiB adj @ 23.3 TB/s.
// ---------------------------------------------------------------------------
__global__ __launch_bounds__(128) void gcn_main_kernel(
    const float* __restrict__ adj, const _Float16* __restrict__ Pt,
    const float* __restrict__ bias, float* __restrict__ out)
{
  const int lane  = threadIdx.x & 31;
  const int wave  = threadIdx.x >> 5;
  const int half  = lane >> 4;             // 0: lanes 0-15, 1: lanes 16-31
  const int l15   = lane & 15;
  const int khalf = half * 8;              // K sub-offset per A/B layout

  const int rowBlock = blockIdx.x * 64;
  const int mrow     = rowBlock + wave * 16 + l15;   // this lane's A row
  const float* __restrict__ arow = adj + (size_t)mrow * NNODES;

  // B base pointers: lane l15 is column n = j*16 + l15 of the B matrix,
  // stored as Pt[n][k] with k contiguous (transposed projection).
  const _Float16* bbase[8];
#pragma unroll
  for (int j = 0; j < 8; ++j)
    bbase[j] = Pt + (size_t)(j * 16 + l15) * NNODES + khalf;

  v8f acc[8] = {};

  // ---- Prologue: load chunk 0 fragments
  v4f p0 = __builtin_nontemporal_load((const v4f*)(arow + khalf));
  v4f p1 = __builtin_nontemporal_load((const v4f*)(arow + khalf + 4));
  v4f p2 = __builtin_nontemporal_load((const v4f*)(arow + khalf + 16));
  v4f p3 = __builtin_nontemporal_load((const v4f*)(arow + khalf + 20));
  BFrag bf[8];
#pragma unroll
  for (int j = 0; j < 8; ++j) {
    bf[j].h[0] = *(const v8h*)(bbase[j]);
    bf[j].h[1] = *(const v8h*)(bbase[j] + 16);
  }
  v16h afA = cvt_afrag(p0, p1, p2, p3);
  v16h afB;

  // ---- Main loop: 512 chunks of K=32, unrolled 2x for A ping-pong.
  // Prefetch chunk index wraps via & (NNODES-1): tail issues harmless
  // in-bounds dummy loads, keeping the loop branch-free.
  for (int kc = 0; kc < NNODES; kc += 64) {
    gemm_step(arow, bbase, (kc + 32) & (NNODES - 1), khalf, acc, afA, afB, bf);
    gemm_step(arow, bbase, (kc + 64) & (NNODES - 1), khalf, acc, afB, afA, bf);
  }

  // ---- Epilogue. C/D layout: lane -> N = l15; VGPR r -> M = r + 8*half.
  const int rbase = rowBlock + wave * 16 + half * 8;
#pragma unroll
  for (int j = 0; j < 8; ++j) {
    const int n = j * 16 + l15;
    const float bn = bias[n];
#pragma unroll
    for (int r = 0; r < 8; ++r) {
      __builtin_nontemporal_store(acc[j][r] + bn,
                                  &out[(size_t)(rbase + r) * FDIM + n]);
    }
  }
}

// ---------------------------------------------------------------------------
extern "C" void kernel_launch(void* const* d_in, const int* in_sizes, int n_in,
                              void* d_out, int out_size, void* d_ws, size_t ws_size,
                              hipStream_t stream) {
  const float* adj  = (const float*)d_in[0];   // [N, N] fp32
  const float* feat = (const float*)d_in[1];   // [N, 128] fp32
  const float* W    = (const float*)d_in[2];   // [128, 128] fp32
  const float* b    = (const float*)d_in[3];   // [128] fp32
  float* out = (float*)d_out;                  // [N, 128] fp32

  _Float16* Pt = (_Float16*)d_ws;              // [128][N] fp16 = 4 MB scratch

  gcn_proj_kernel<<<NNODES, FDIM, 0, stream>>>(feat, W, Pt);
  gcn_main_kernel<<<NNODES / 64, 128, 0, stream>>>(adj, Pt, b, out);
}
